// SupConLossBruteForce_4518305595515
// MI455X (gfx1250) — compile-verified, matrix operand order checked by reading, not run on hardware
//
#include <hip/hip_runtime.h>
#include <hip/hip_bf16.h>

typedef __attribute__((ext_vector_type(16))) _Float16 v16h;
typedef __attribute__((ext_vector_type(8)))  _Float16 v8h;
typedef __attribute__((ext_vector_type(4)))  _Float16 v4h;
typedef __attribute__((ext_vector_type(8)))  float    v8f;

#define SN 8192
#define SD 128
#define JSPLIT 16
#define NTILE (SN / 16)

// ---------------------------------------------------------------------------
// Kernel 1: L2-normalize rows (f32 -> f16). One wave per row; lane handles 4
// floats, wave-wide shfl_xor allreduce for the squared norm.
// ---------------------------------------------------------------------------
__global__ __launch_bounds__(256) void supcon_normalize(const float* __restrict__ feat,
                                                        _Float16* __restrict__ fh) {
  const int lane = threadIdx.x & 31;
  const int wv   = threadIdx.x >> 5;
  const int row  = blockIdx.x * 8 + wv;
  const float* fr = feat + (size_t)row * SD;
  const float4 v = *(const float4*)(fr + lane * 4);
  float ss = v.x * v.x + v.y * v.y + v.z * v.z + v.w * v.w;
#pragma unroll
  for (int m = 1; m < 32; m <<= 1) ss += __shfl_xor(ss, m, 32);
  const float scale = 1.0f / fmaxf(sqrtf(ss), 1e-12f);
  v4h h;
  h[0] = (_Float16)(v.x * scale);
  h[1] = (_Float16)(v.y * scale);
  h[2] = (_Float16)(v.z * scale);
  h[3] = (_Float16)(v.w * scale);
  *(v4h*)(fh + (size_t)row * SD + lane * 4) = h;
}

// Load one 16x32 f16 WMMA fragment for this lane: row = tile row (lane&15),
// K-chunks per the CDNA5 A/B 16-bit layout (half = lane>>4).
__device__ __forceinline__ v16h load_frag(const _Float16* __restrict__ rowp, int K0, int half) {
  const v8h lo = *(const v8h*)(rowp + K0 + 8 * half);
  const v8h hi = *(const v8h*)(rowp + K0 + 16 + 8 * half);
  v16h r;
#pragma unroll
  for (int i = 0; i < 8; ++i) { r[i] = lo[i]; r[i + 8] = hi[i]; }
  return r;
}

struct BTile {
  v16h b0, b1, b2, b3;
  int  gj;   // this lane's global column index
  int  lj;   // label of that column
};

__device__ __forceinline__ void load_btile(const _Float16* __restrict__ fh,
                                           const int* __restrict__ labels,
                                           int jbase, int lm, int half, BTile& t) {
  const _Float16* brow = fh + (size_t)(jbase + lm) * SD;
  t.b0 = load_frag(brow, 0, half);
  t.b1 = load_frag(brow, 32, half);
  t.b2 = load_frag(brow, 64, half);
  t.b3 = load_frag(brow, 96, half);
  t.gj = jbase + lm;
  t.lj = labels[t.gj];
}

struct Acc {
  float E[8], P[8], C[8];
};

__device__ __forceinline__ void process_tile(const v16h& a0, const v16h& a1,
                                             const v16h& a2, const v16h& a3,
                                             const BTile& t, const int* li,
                                             int ibase, int half, Acc& acc) {
  v8f c = {0.f, 0.f, 0.f, 0.f, 0.f, 0.f, 0.f, 0.f};
  c = __builtin_amdgcn_wmma_f32_16x16x32_f16(false, a0, false, t.b0, (short)0, c, false, false);
  c = __builtin_amdgcn_wmma_f32_16x16x32_f16(false, a1, false, t.b1, (short)0, c, false, false);
  c = __builtin_amdgcn_wmma_f32_16x16x32_f16(false, a2, false, t.b2, (short)0, c, false, false);
  c = __builtin_amdgcn_wmma_f32_16x16x32_f16(false, a3, false, t.b3, (short)0, c, false, false);

  // C/D layout: VGPR r -> (M = r + 8*half, N = lane&15)
#pragma unroll
  for (int r = 0; r < 8; ++r) {
    const float s  = c[r];                 // S_ij (tau == 1)
    const int   gi = ibase + r + 8 * half;
    const bool  nd = (gi != t.gj);
    const float e  = __expf(s);
    acc.E[r] += nd ? e : 0.f;
    const bool pos = nd && (li[r] == t.lj);
    acc.P[r] += pos ? s : 0.f;
    acc.C[r] += pos ? 1.f : 0.f;
  }
}

// ---------------------------------------------------------------------------
// Kernel 2: tiled f fT with fused exp-sum / positive-sum / positive-count.
// Block = 8 waves; each wave owns (itile, slice) and strides j-tiles by JSPLIT.
// The j-loop is unrolled 2x with explicitly alternating B buffers: each buffer
// is loaded in place while the other feeds the WMMA chain — no ping-pong
// register copies, and next-tile loads overlap WMMA + exp/mask work.
// ---------------------------------------------------------------------------
__global__ __launch_bounds__(256) void supcon_tiles(const _Float16* __restrict__ fh,
                                                    const int* __restrict__ labels,
                                                    float* __restrict__ expP,
                                                    float* __restrict__ posP,
                                                    float* __restrict__ cntP) {
  const int lane  = threadIdx.x & 31;
  const int wv    = threadIdx.x >> 5;
  const int lm    = lane & 15;
  const int half  = lane >> 4;
  const int itile = blockIdx.x;
  const int slice = blockIdx.y * 8 + wv;
  const int ibase = itile * 16;

  const _Float16* arow = fh + (size_t)(ibase + lm) * SD;
  const v16h a0 = load_frag(arow, 0, half);
  const v16h a1 = load_frag(arow, 32, half);
  const v16h a2 = load_frag(arow, 64, half);
  const v16h a3 = load_frag(arow, 96, half);

  int li[8];
#pragma unroll
  for (int r = 0; r < 8; ++r) li[r] = labels[ibase + r + 8 * half];

  Acc acc;
#pragma unroll
  for (int r = 0; r < 8; ++r) { acc.E[r] = 0.f; acc.P[r] = 0.f; acc.C[r] = 0.f; }

  // NTILE/JSPLIT == 32 iterations per wave (even), identical for every slice.
  BTile t0, t1;
  load_btile(fh, labels, slice * 16, lm, half, t0);

  for (int jt = slice; jt < NTILE; jt += 2 * JSPLIT) {
    // jt+JSPLIT is always in range (even trip count).
    load_btile(fh, labels, (jt + JSPLIT) * 16, lm, half, t1);   // prefetch odd
    process_tile(a0, a1, a2, a3, t0, li, ibase, half, acc);     // consume even

    const int jn2 = jt + 2 * JSPLIT;
    if (jn2 < NTILE)
      load_btile(fh, labels, jn2 * 16, lm, half, t0);           // prefetch even
    process_tile(a0, a1, a2, a3, t1, li, ibase, half, acc);     // consume odd
  }

  // Reduce across the 16 N-lanes within each half-wave (masks <16 stay in-half).
#pragma unroll
  for (int r = 0; r < 8; ++r) {
#pragma unroll
    for (int m = 1; m < 16; m <<= 1) {
      acc.E[r] += __shfl_xor(acc.E[r], m, 32);
      acc.P[r] += __shfl_xor(acc.P[r], m, 32);
      acc.C[r] += __shfl_xor(acc.C[r], m, 32);
    }
  }

  if (lm == 0) {  // lane 0 writes rows 0..7, lane 16 writes rows 8..15
#pragma unroll
    for (int r = 0; r < 8; ++r) {
      const int row = ibase + r + 8 * half;
      const int idx = row * JSPLIT + slice;
      expP[idx] = acc.E[r];
      posP[idx] = acc.P[r];
      cntP[idx] = acc.C[r];
    }
  }
}

// ---------------------------------------------------------------------------
// Kernel 3: single block, fixed-order reduction -> deterministic scalar.
// ---------------------------------------------------------------------------
__global__ __launch_bounds__(1024) void supcon_finalize(const float* __restrict__ expP,
                                                        const float* __restrict__ posP,
                                                        const float* __restrict__ cntP,
                                                        float* __restrict__ out) {
  __shared__ float red[1024];
  float local = 0.f;
  for (int row = threadIdx.x; row < SN; row += 1024) {
    float se = 0.f, sp = 0.f, sc = 0.f;
    for (int s = 0; s < JSPLIT; ++s) {
      se += expP[row * JSPLIT + s];
      sp += posP[row * JSPLIT + s];
      sc += cntP[row * JSPLIT + s];
    }
    const float lt = __logf(se);
    local += (sc * lt - sp) / (sc + 1e-5f);
  }
  red[threadIdx.x] = local;
  __syncthreads();
  for (int off = 512; off > 0; off >>= 1) {
    if (threadIdx.x < (unsigned)off) red[threadIdx.x] += red[threadIdx.x + off];
    __syncthreads();
  }
  if (threadIdx.x == 0) out[0] = red[0] / (float)SN;
}

extern "C" void kernel_launch(void* const* d_in, const int* in_sizes, int n_in,
                              void* d_out, int out_size, void* d_ws, size_t ws_size,
                              hipStream_t stream) {
  const float* feat   = (const float*)d_in[0];
  const int*   labels = (const int*)d_in[1];

  char* ws = (char*)d_ws;
  _Float16* fh   = (_Float16*)ws;                                  // 2 MB
  float*    expP = (float*)(ws + (size_t)SN * SD * sizeof(_Float16));
  float*    posP = expP + (size_t)SN * JSPLIT;                     // 512 KB each
  float*    cntP = posP + (size_t)SN * JSPLIT;

  supcon_normalize<<<SN / 8, 256, 0, stream>>>(feat, fh);
  supcon_tiles<<<dim3(NTILE, JSPLIT / 8), 256, 0, stream>>>(fh, labels, expP, posP, cntP);
  supcon_finalize<<<1, 1024, 0, stream>>>(expP, posP, cntP, (float*)d_out);
}
